// TurboLinear_13056700579867
// MI455X (gfx1250) — matrix operand order repaired and checked by reading, not compiled, not run on hardware
//
#include <hip/hip_runtime.h>

#define BDIM 4096   // batch rows of x / out
#define MDIM 4096   // output columns (rows of W)
#define KDIM 4096   // reduction dim
#define KW   (KDIM / 2)   // packed int32 words per W row
#define PNUM 32768

#define BT   128    // block tile: batch rows
#define MT   256    // block tile: output cols
#define KT   32     // K per stage (WMMA depth)
#define NSTAGE (KDIM / KT)
#define LDSS 40     // padded LDS row stride in ushorts (80B = 20 banks -> conflict free b128 reads)

typedef __attribute__((ext_vector_type(16))) __bf16 v16bf;
typedef __attribute__((ext_vector_type(8)))  float  v8f;

union FragBF { uint4 q[2]; v16bf v; };
union AccU   { v8f v; float f[8]; };

// round-to-nearest-even fp32 -> bf16 (matches jnp astype(bfloat16))
__device__ __forceinline__ unsigned bf16r(float f) {
    unsigned u = __float_as_uint(f);
    return (u + 0x7FFFu + ((u >> 16) & 1u)) >> 16;
}
__device__ __forceinline__ unsigned pack2bf(float a, float b) {
    return bf16r(a) | (bf16r(b) << 16);
}

// decode one packed word (two 12-bit codes) -> two bf16 (lo code -> bits[15:0])
// value = +/- (1 + m/128) * 2^(e + base_exp - 16) is exactly bf16:
// bits = s<<15 | ((code & 0x7FF) + (base_exp+111)<<7)
__device__ __forceinline__ unsigned decode2(unsigned w, unsigned ebias7) {
    unsigned lo = w & 0xFFFu;
    unsigned hi = (w >> 12) & 0xFFFu;
    unsigned blo = ((lo & 0x800u) << 4) | ((lo & 0x7FFu) + ebias7);
    unsigned bhi = ((hi & 0x800u) << 4) | ((hi & 0x7FFu) + ebias7);
    return blo | (bhi << 16);
}

__global__ __launch_bounds__(256)
void turbo_gemm_kernel(const float* __restrict__ x,
                       const int*   __restrict__ packed,
                       const float* __restrict__ bias,
                       const int*   __restrict__ base_exp,
                       float*       __restrict__ out)
{
    // double-buffered stage tiles: X 128x32 bf16, W 256x32 bf16 (rows padded)
    __shared__ __align__(16) unsigned short Xs[2][BT][LDSS];
    __shared__ __align__(16) unsigned short Ws[2][MT][LDSS];

    const int t    = threadIdx.x;
    const int lane = t & 31;
    const int wid  = t >> 5;
    const unsigned ebias7 = (unsigned)(base_exp[0] + 111) << 7;

    const int rB = blockIdx.y * BT;     // batch-row base
    const int rM = blockIdx.x * MT;     // output-col base (W row base)

    // cooperative global-load coordinates
    const int xrow = t >> 3;            // + i*32, i=0..3  : 128 rows of X
    const int xoff = (t & 7) * 4;       // float index within 32-wide K chunk
    const int wrow = t >> 2;            // + i*64, i=0..3  : 256 rows of W
    const int woff = (t & 3) * 4;       // word index within 16-word chunk

    // wave's 64x64 sub-tile: 8 waves as 2 (B) x 4 (M)
    const int wr = (wid >> 2) * 64;     // row offset (B dim)
    const int wc = (wid & 3) * 64;      // col offset (M dim)
    const int lr = lane & 15;
    const bool hihalf = lane >= 16;
    const int aoff = hihalf ? 8  : 0;   // A frag: K halves {0..7,16..23} / {8..15,24..31}
    const int boff = hihalf ? 8  : 0;   // B frag: K halves {0..7,16..23} / {8..15,24..31}

    AccU acc[4][4];
    #pragma unroll
    for (int i = 0; i < 4; ++i)
        #pragma unroll
        for (int j = 0; j < 4; ++j)
            #pragma unroll
            for (int e = 0; e < 8; ++e) acc[i][j].f[e] = 0.0f;

    float4 xreg[4];
    int4   wreg[4];

    // ---- prologue: stage 0 -> buf0, issue stage 1 loads ----
    #pragma unroll
    for (int i = 0; i < 4; ++i)
        xreg[i] = *(const float4*)&x[(size_t)(rB + xrow + i * 32) * KDIM + xoff];
    #pragma unroll
    for (int i = 0; i < 4; ++i)
        wreg[i] = *(const int4*)&packed[(size_t)(rM + wrow + i * 64) * KW + woff];

    #pragma unroll
    for (int i = 0; i < 4; ++i) {
        uint2 p;
        p.x = pack2bf(xreg[i].x, xreg[i].y);
        p.y = pack2bf(xreg[i].z, xreg[i].w);
        *(uint2*)&Xs[0][xrow + i * 32][xoff] = p;
    }
    #pragma unroll
    for (int i = 0; i < 4; ++i) {
        uint4 q;
        q.x = decode2((unsigned)wreg[i].x, ebias7);
        q.y = decode2((unsigned)wreg[i].y, ebias7);
        q.z = decode2((unsigned)wreg[i].z, ebias7);
        q.w = decode2((unsigned)wreg[i].w, ebias7);
        *(uint4*)&Ws[0][wrow + i * 64][woff * 2] = q;
    }
    #pragma unroll
    for (int i = 0; i < 4; ++i)
        xreg[i] = *(const float4*)&x[(size_t)(rB + xrow + i * 32) * KDIM + KT + xoff];
    #pragma unroll
    for (int i = 0; i < 4; ++i)
        wreg[i] = *(const int4*)&packed[(size_t)(rM + wrow + i * 64) * KW + (KT / 2) + woff];
    __syncthreads();

    // ---- main loop: one barrier per stage (double buffer) ----
    for (int kt = 0; kt < NSTAGE; ++kt) {
        const int cur = kt & 1;
        const int nxt = cur ^ 1;

        // store stage kt+1 (loaded last iteration) into the other buffer
        if (kt + 1 < NSTAGE) {
            #pragma unroll
            for (int i = 0; i < 4; ++i) {
                uint2 p;
                p.x = pack2bf(xreg[i].x, xreg[i].y);
                p.y = pack2bf(xreg[i].z, xreg[i].w);
                *(uint2*)&Xs[nxt][xrow + i * 32][xoff] = p;
            }
            #pragma unroll
            for (int i = 0; i < 4; ++i) {
                uint4 q;
                q.x = decode2((unsigned)wreg[i].x, ebias7);
                q.y = decode2((unsigned)wreg[i].y, ebias7);
                q.z = decode2((unsigned)wreg[i].z, ebias7);
                q.w = decode2((unsigned)wreg[i].w, ebias7);
                *(uint4*)&Ws[nxt][wrow + i * 64][woff * 2] = q;
            }
        }
        // issue global loads for stage kt+2 (in flight across this compute stage)
        if (kt + 2 < NSTAGE) {
            const int k0 = (kt + 2) * KT;
            const int w0 = (kt + 2) * (KT / 2);
            #pragma unroll
            for (int i = 0; i < 4; ++i)
                xreg[i] = *(const float4*)&x[(size_t)(rB + xrow + i * 32) * KDIM + k0 + xoff];
            #pragma unroll
            for (int i = 0; i < 4; ++i)
                wreg[i] = *(const int4*)&packed[(size_t)(rM + wrow + i * 64) * KW + w0 + woff];
        }

        // fragments from LDS, 16 WMMAs
        FragBF a[4], b[4];
        #pragma unroll
        for (int i = 0; i < 4; ++i) {
            const unsigned short* p = &Xs[cur][wr + i * 16 + lr][aoff];
            a[i].q[0] = *(const uint4*)p;
            a[i].q[1] = *(const uint4*)(p + 16);
        }
        #pragma unroll
        for (int j = 0; j < 4; ++j) {
            const unsigned short* p = &Ws[cur][wc + j * 16 + lr][boff];
            b[j].q[0] = *(const uint4*)p;
            b[j].q[1] = *(const uint4*)(p + 16);
        }
        #pragma unroll
        for (int i = 0; i < 4; ++i)
            #pragma unroll
            for (int j = 0; j < 4; ++j)
                acc[i][j].v = __builtin_amdgcn_wmma_f32_16x16x32_bf16(
                    false, a[i].v, false, b[j].v,
                    (short)0, acc[i][j].v, false, false);

        __syncthreads();
    }

    // ---- epilogue: C layout (VGPR e, lane l) -> row = e + (l>=16 ? 8 : 0), col = l%16 ----
    #pragma unroll
    for (int j = 0; j < 4; ++j) {
        const int col = rM + wc + j * 16 + lr;
        const float bj = bias[col];
        #pragma unroll
        for (int i = 0; i < 4; ++i) {
            #pragma unroll
            for (int e = 0; e < 8; ++e) {
                const int row = rB + wr + i * 16 + e + (hihalf ? 8 : 0);
                out[(size_t)row * MDIM + col] = acc[i][j].f[e] + bj;
            }
        }
    }
}

// out[b, rows[p]] += (correct[p] - wrong[p]) * bf16(x[b, cols[p]])
__global__ __launch_bounds__(256)
void turbo_patch_kernel(const float* __restrict__ x,
                        const int*   __restrict__ prow,
                        const int*   __restrict__ pcol,
                        const float* __restrict__ pcor,
                        const float* __restrict__ pwr,
                        float*       __restrict__ out)
{
    const int p = blockIdx.x * blockDim.x + threadIdx.x;   // 0..PNUM-1
    const int b = blockIdx.y;                              // 0..BDIM-1
    const int r = prow[p];
    const int c = pcol[p];
    const float d = pcor[p] - pwr[p];
    // bf16-round x then expand back to f32 (matches reference xb)
    unsigned u = __float_as_uint(x[(size_t)b * KDIM + c]);
    u = (u + 0x7FFFu + ((u >> 16) & 1u)) & 0xFFFF0000u;
    const float xb = __uint_as_float(u);
    atomicAdd(&out[(size_t)b * MDIM + r], d * xb);
}

extern "C" void kernel_launch(void* const* d_in, const int* in_sizes, int n_in,
                              void* d_out, int out_size, void* d_ws, size_t ws_size,
                              hipStream_t stream) {
    const float* x        = (const float*)d_in[0];
    const int*   packed   = (const int*)  d_in[1];
    const int*   prow     = (const int*)  d_in[2];
    const int*   pcol     = (const int*)  d_in[3];
    const float* pcorrect = (const float*)d_in[4];
    const float* pwrong   = (const float*)d_in[5];
    const float* bias     = (const float*)d_in[6];
    const int*   base_exp = (const int*)  d_in[7];
    float* out = (float*)d_out;

    dim3 ggrid(MDIM / MT, BDIM / BT);       // 16 x 32
    turbo_gemm_kernel<<<ggrid, 256, 0, stream>>>(x, packed, bias, base_exp, out);

    dim3 pgrid(PNUM / 256, BDIM);           // 128 x 4096
    turbo_patch_kernel<<<pgrid, 256, 0, stream>>>(x, prow, pcol, pcorrect, pwrong, out);
}
